// GraphIsomorphismNetwork_24816321036833
// MI455X (gfx1250) — matrix operand order repaired and checked by reading, not compiled, not run on hardware
//
#include <hip/hip_runtime.h>
#include <hip/hip_bf16.h>

typedef __attribute__((ext_vector_type(16))) __bf16 v16bf;
typedef __attribute__((ext_vector_type(8)))  __bf16 v8bf;
typedef __attribute__((ext_vector_type(8)))  float  v8f;

#define N_NODES 50000
#define N_EDGES 800000
#define N_FEAT  128
#define DDIM    64
#define NLAYERS 6
#define NGRAPHS 50
#define BN_EPS  1e-5f

// ---------------------------------------------------------------- utilities
__global__ void k_zero(float* __restrict__ p, int n) {
    int i = blockIdx.x * blockDim.x + threadIdx.x;
    if (i < n) p[i] = 0.0f;
}

// ------------------------------------------------- WMMA GEMM: [M,K] @ [K,64]
// One wave computes a 16x64 output tile with 4 accumulators of
// v_wmma_f32_16x16x32_bf16.
//  * Weights enter as fp32 and are converted+swizzled into LDS so each lane's
//    B fragment is 16 *contiguous* bf16 (layout: b[j] = W[k0+half*8+j][n], +16 for j>=8)
//    -> fragment load = 2x ds_load_b128.
//  * A fragment is two contiguous 8-float runs (a[j]=A[m][k0+half*8+j], +16)
//    -> 4x global_load_b128 + packed cvt.
//  * Output goes through a per-wave LDS tile so global stores are coalesced
//    float4 (no barrier needed: LDS ops within a wave are in-order).
__global__ __launch_bounds__(256)
void k_gemm_wmma(const float* __restrict__ A, const float* __restrict__ Wf,
                 const float* __restrict__ bias, float* __restrict__ C,
                 int M, int K, int relu) {
    __shared__ __bf16 sB[N_FEAT * DDIM];      // swizzled fragments, max 16 KB
    __shared__ float  sO[8][16][DDIM];        // per-wave store staging, 32 KB
    const int tid = threadIdx.x;

    // stage + convert + swizzle weights:
    // half-fragment h -> (kc,nt,lane,hi); source = W[kc*32+hi*16+half*8+r][nt*16+lane%16]
    const int nhalf = (K * DDIM) >> 3;
    for (int h = tid; h < nhalf; h += 256) {
        int l    = (h >> 1) & 31;
        int nt   = (h >> 6) & 3;
        int kc   = h >> 8;
        int krow = kc * 32 + (l >> 4) * 8 + (h & 1) * 16;
        int n    = nt * 16 + (l & 15);
        v8bf pk;
#pragma unroll
        for (int r = 0; r < 8; ++r)
            pk[r] = (__bf16)Wf[(krow + r) * DDIM + n];
        *(v8bf*)(sB + h * 8) = pk;
    }
    __syncthreads();

    const int wave = tid >> 5;
    const int lane = tid & 31;
    const int mloc = lane & 15;
    const int half = lane >> 4;
    const int tile = blockIdx.x * 8 + wave;
    if (tile >= (M >> 4)) return;             // wave-uniform; EXEC all-ones inside

    v8f acc[4] = {v8f{}, v8f{}, v8f{}, v8f{}};
    const float* arow = A + (size_t)(tile * 16 + mloc) * K;

    for (int k0 = 0; k0 < K; k0 += 32) {
        const float4* ap = (const float4*)(arow + k0 + half * 8);
        float4 f0 = ap[0], f1 = ap[1];        // K = k0+half*8    .. +7   -> a[0..7]
        float4 f2 = ap[4], f3 = ap[5];        // K = k0+16+half*8 .. +7   -> a[8..15]
        v16bf a;
        a[0]  = (__bf16)f0.x; a[1]  = (__bf16)f0.y; a[2]  = (__bf16)f0.z; a[3]  = (__bf16)f0.w;
        a[4]  = (__bf16)f1.x; a[5]  = (__bf16)f1.y; a[6]  = (__bf16)f1.z; a[7]  = (__bf16)f1.w;
        a[8]  = (__bf16)f2.x; a[9]  = (__bf16)f2.y; a[10] = (__bf16)f2.z; a[11] = (__bf16)f2.w;
        a[12] = (__bf16)f3.x; a[13] = (__bf16)f3.y; a[14] = (__bf16)f3.z; a[15] = (__bf16)f3.w;

        const v16bf* bfrag = (const v16bf*)(sB + (size_t)(k0 >> 5) * 4 * 32 * 16);
#pragma unroll
        for (int nt = 0; nt < 4; ++nt) {
            v16bf b = bfrag[nt * 32 + lane];  // 2x ds_load_b128
            acc[nt] = __builtin_amdgcn_wmma_f32_16x16x32_bf16(
                false, a, false, b, (short)0, acc[nt], false, false);
        }
    }

    // epilogue: bias (+ReLU) -> per-wave LDS tile -> coalesced float4 stores
    float (*so)[DDIM] = sO[wave];
#pragma unroll
    for (int nt = 0; nt < 4; ++nt) {
        int n = nt * 16 + mloc;
        float bv = bias[n];
#pragma unroll
        for (int r = 0; r < 8; ++r) {
            float v = acc[nt][r] + bv;
            if (relu) v = fmaxf(v, 0.0f);
            so[half * 8 + r][n] = v;          // C/D layout: row = r + 8*half
        }
    }
    float4* crow = (float4*)(C + (size_t)tile * 16 * DDIM);
#pragma unroll
    for (int it = 0; it < 8; ++it) {
        int q   = it * 32 + lane;             // float4 id in 16x64 tile
        int row = q >> 4, seg = q & 15;
        crow[q] = *(const float4*)(&so[row][seg * 4]);
    }
}

// ------------------------------------------------ edge scatter: weighted add
// 4 edges per block; edge metadata staged once in LDS.
__global__ __launch_bounds__(256)
void k_scatter(const float* __restrict__ x, const float* __restrict__ ew,
               const int* __restrict__ src, const int* __restrict__ dst,
               float* __restrict__ aggr) {
    __shared__ int   s_src[4], s_dst[4];
    __shared__ float s_w[4];
    const int tid = threadIdx.x;
    const int e0  = blockIdx.x * 4;           // N_EDGES % 4 == 0
    if (tid < 4) {
        s_src[tid] = src[e0 + tid];
        s_dst[tid] = dst[e0 + tid];
        s_w[tid]   = ew[e0 + tid];
    }
    __syncthreads();
    const int es = tid >> 6;
    const int d  = tid & 63;
    float v = s_w[es] * x[(size_t)s_src[es] * DDIM + d];
    unsafeAtomicAdd(&aggr[(size_t)s_dst[es] * DDIM + d], v); // global_atomic_add_f32
}

// -------------------------------------------- per-column sum / sum-of-squares
__global__ __launch_bounds__(256)
void k_colstats(const float* __restrict__ X, float* __restrict__ stats, int M) {
    __shared__ float s1[256], s2[256];
    int tid = threadIdx.x;
    int col = tid & 63;
    int rg  = tid >> 6;                       // 0..3
    float sum = 0.0f, ss = 0.0f;
    for (int row = blockIdx.x * 4 + rg; row < M; row += gridDim.x * 4) {
        float v = X[(size_t)row * DDIM + col];
        sum += v; ss += v * v;
    }
    s1[tid] = sum; s2[tid] = ss;
    __syncthreads();
    if (tid < 64) {
        float t1 = s1[tid] + s1[tid + 64] + s1[tid + 128] + s1[tid + 192];
        float t2 = s2[tid] + s2[tid + 64] + s2[tid + 128] + s2[tid + 192];
        unsafeAtomicAdd(&stats[col], t1);
        unsafeAtomicAdd(&stats[64 + col], t2);
    }
}

// --------------------------------------------------- BN(train) + ReLU inplace
__global__ void k_bn_relu(float* __restrict__ X, const float* __restrict__ stats,
                          const float* __restrict__ gamma, const float* __restrict__ beta,
                          int M) {
    size_t i = (size_t)blockIdx.x * blockDim.x + threadIdx.x;
    if (i >= (size_t)M * DDIM) return;
    int col = (int)(i & 63);
    float mean = stats[col] / (float)M;
    float var  = stats[64 + col] / (float)M - mean * mean;
    float inv  = rsqrtf(var + BN_EPS);
    float v = gamma[col] * (X[i] - mean) * inv + beta[col];
    X[i] = fmaxf(v, 0.0f);
}

// --------------------------- BN + ReLU -> new activations + global add pool
__global__ void k_bn_relu_pool(const float* __restrict__ X, float* __restrict__ Xout,
                               const float* __restrict__ stats,
                               const float* __restrict__ gamma, const float* __restrict__ beta,
                               const int* __restrict__ batch, float* __restrict__ pooled,
                               int layer, int M) {
    size_t i = (size_t)blockIdx.x * blockDim.x + threadIdx.x;
    if (i >= (size_t)M * DDIM) return;
    int row = (int)(i >> 6);
    int col = (int)(i & 63);
    float mean = stats[col] / (float)M;
    float var  = stats[64 + col] / (float)M - mean * mean;
    float inv  = rsqrtf(var + BN_EPS);
    float v = gamma[col] * (X[i] - mean) * inv + beta[col];
    v = fmaxf(v, 0.0f);
    Xout[i] = v;
    int g = batch[row];
    // pooled stored as [G, 6, 64] == the reference's [G, 6*64] concat layout
    unsafeAtomicAdd(&pooled[((size_t)g * NLAYERS + layer) * DDIM + col], v);
}

// ------------------------------------------------ tiny dense for post-MLP
__global__ void k_dense(const float* __restrict__ A, const float* __restrict__ W,
                        const float* __restrict__ b, float* __restrict__ O,
                        int G, int K, int N, int relu) {
    int idx = blockIdx.x * blockDim.x + threadIdx.x;
    if (idx >= G * N) return;
    int g = idx / N, n = idx % N;
    float s = b[n];
    for (int k = 0; k < K; ++k) s += A[g * K + k] * W[k * N + n];
    if (relu) s = fmaxf(s, 0.0f);
    O[idx] = s;
}

// ---------------------------------------------------------------------------
static inline int cdiv(long a, long b) { return (int)((a + b - 1) / b); }

extern "C" void kernel_launch(void* const* d_in, const int* in_sizes, int n_in,
                              void* d_out, int out_size, void* d_ws, size_t ws_size,
                              hipStream_t stream) {
    const float* x        = (const float*)d_in[0];
    const float* ew       = (const float*)d_in[1];
    const float* pre_W1_0 = (const float*)d_in[2];
    const float* pre_b1_0 = (const float*)d_in[3];
    const float* pre_W2_0 = (const float*)d_in[4];
    const float* pre_b2_0 = (const float*)d_in[5];
    const float* pre_W1   = (const float*)d_in[6];
    const float* pre_b1   = (const float*)d_in[7];
    const float* pre_W2   = (const float*)d_in[8];
    const float* pre_b2   = (const float*)d_in[9];
    const float* g_W1     = (const float*)d_in[10];
    const float* g_b1     = (const float*)d_in[11];
    const float* g_gamma1 = (const float*)d_in[12];
    const float* g_beta1  = (const float*)d_in[13];
    const float* g_W2     = (const float*)d_in[14];
    const float* g_b2     = (const float*)d_in[15];
    const float* g_gamma2 = (const float*)d_in[16];
    const float* g_beta2  = (const float*)d_in[17];
    const float* post0_W1 = (const float*)d_in[18];
    const float* post0_b1 = (const float*)d_in[19];
    const float* post0_W2 = (const float*)d_in[20];
    const float* post0_b2 = (const float*)d_in[21];
    const float* post1_W1 = (const float*)d_in[22];
    const float* post1_b1 = (const float*)d_in[23];
    const float* post1_W2 = (const float*)d_in[24];
    const float* post1_b2 = (const float*)d_in[25];
    const float* post2_W1 = (const float*)d_in[26];
    const float* post2_b1 = (const float*)d_in[27];
    const float* post2_W2 = (const float*)d_in[28];
    const float* post2_b2 = (const float*)d_in[29];
    const int*   eidx     = (const int*)d_in[30];
    const int*   batch    = (const int*)d_in[31];
    const int* esrc = eidx;
    const int* edst = eidx + N_EDGES;
    float* out = (float*)d_out;

    // -------- workspace layout
    char* ws = (char*)d_ws;
    const size_t NB = (size_t)N_NODES * DDIM * sizeof(float);   // 12.8 MB
    float* buf0   = (float*)ws;  ws += NB;
    float* buf1   = (float*)ws;  ws += NB;
    float* buf2   = (float*)ws;  ws += NB;                      // aggr / temp
    float* pooled = (float*)ws;  ws += (size_t)NGRAPHS * NLAYERS * DDIM * sizeof(float);
    float* stats  = (float*)ws;  ws += 128 * sizeof(float);
    float* ha     = (float*)ws;  ws += NGRAPHS * DDIM * sizeof(float);
    float* hb     = (float*)ws;  ws += NGRAPHS * DDIM * sizeof(float);

    const int T = 256;
    const int gemm_blocks = cdiv(N_NODES / 16, 8);              // 391
    const size_t NE = (size_t)N_NODES * DDIM;                   // 3.2M
    const int ne_blocks = cdiv((long)NE, T);

    // -------- pre-MLP (6 WMMA GEMMs, ReLU fused, fp32 weights converted in-kernel)
    k_gemm_wmma<<<gemm_blocks, T, 0, stream>>>(x,    pre_W1_0, pre_b1_0, buf0, N_NODES, N_FEAT, 1);
    k_gemm_wmma<<<gemm_blocks, T, 0, stream>>>(buf0, pre_W2_0, pre_b2_0, buf1, N_NODES, DDIM, 1);
    for (int i = 0; i < 2; ++i) {
        k_gemm_wmma<<<gemm_blocks, T, 0, stream>>>(buf1, pre_W1 + (size_t)i * DDIM * DDIM,
                                                   pre_b1 + i * DDIM, buf0, N_NODES, DDIM, 1);
        k_gemm_wmma<<<gemm_blocks, T, 0, stream>>>(buf0, pre_W2 + (size_t)i * DDIM * DDIM,
                                                   pre_b2 + i * DDIM, buf1, N_NODES, DDIM, 1);
    }
    // activations now in buf1

    k_zero<<<cdiv(NGRAPHS * NLAYERS * DDIM, T), T, 0, stream>>>(pooled, NGRAPHS * NLAYERS * DDIM);

    // -------- GIN layers
    for (int k = 0; k < NLAYERS; ++k) {
        k_zero<<<ne_blocks, T, 0, stream>>>(buf2, (int)NE);
        k_scatter<<<N_EDGES / 4, T, 0, stream>>>(buf1, ew, esrc, edst, buf2);

        k_gemm_wmma<<<gemm_blocks, T, 0, stream>>>(buf2, g_W1 + (size_t)k * DDIM * DDIM,
                                                   g_b1 + k * DDIM, buf0, N_NODES, DDIM, 0);
        k_zero<<<1, 128, 0, stream>>>(stats, 128);
        k_colstats<<<200, T, 0, stream>>>(buf0, stats, N_NODES);
        k_bn_relu<<<ne_blocks, T, 0, stream>>>(buf0, stats, g_gamma1 + k * DDIM,
                                               g_beta1 + k * DDIM, N_NODES);

        k_gemm_wmma<<<gemm_blocks, T, 0, stream>>>(buf0, g_W2 + (size_t)k * DDIM * DDIM,
                                                   g_b2 + k * DDIM, buf2, N_NODES, DDIM, 0);
        k_zero<<<1, 128, 0, stream>>>(stats, 128);
        k_colstats<<<200, T, 0, stream>>>(buf2, stats, N_NODES);
        k_bn_relu_pool<<<ne_blocks, T, 0, stream>>>(buf2, buf1, stats, g_gamma2 + k * DDIM,
                                                    g_beta2 + k * DDIM, batch, pooled,
                                                    k, N_NODES);
    }

    // -------- post-MLP on [50, 384] (tiny; plain VALU)
    k_dense<<<cdiv(NGRAPHS * DDIM, T), T, 0, stream>>>(pooled, post0_W1, post0_b1, ha, NGRAPHS, NLAYERS * DDIM, DDIM, 1);
    k_dense<<<cdiv(NGRAPHS * DDIM, T), T, 0, stream>>>(ha, post0_W2, post0_b2, hb, NGRAPHS, DDIM, DDIM, 1);
    k_dense<<<cdiv(NGRAPHS * DDIM, T), T, 0, stream>>>(hb, post1_W1, post1_b1, ha, NGRAPHS, DDIM, DDIM, 1);
    k_dense<<<cdiv(NGRAPHS * DDIM, T), T, 0, stream>>>(ha, post1_W2, post1_b2, hb, NGRAPHS, DDIM, DDIM, 1);
    k_dense<<<cdiv(NGRAPHS * DDIM, T), T, 0, stream>>>(hb, post2_W1, post2_b1, ha, NGRAPHS, DDIM, DDIM, 1);
    k_dense<<<1, 64, 0, stream>>>(ha, post2_W2, post2_b2, out, NGRAPHS, DDIM, 1, 0);
}
